// BertSelfAttnDimensionReduction_80685255623402
// MI455X (gfx1250) — matrix operand order repaired and checked by reading, not compiled, not run on hardware
//
#include <hip/hip_runtime.h>
#include <hip/hip_bf16.h>
#include <math.h>

typedef _Float16 f16;
typedef __attribute__((ext_vector_type(8)))  _Float16 v8h;
typedef __attribute__((ext_vector_type(16))) _Float16 v16h;
typedef __attribute__((ext_vector_type(8)))  float    v8f;

#define SQ    4096
#define SK    4096
#define DIN   768
#define AH    384
#define NH    12
#define HD    32

__device__ __forceinline__ v16h loadv16(const f16* p) {
    v8h lo = *(const v8h*)(p);
    v8h hi = *(const v8h*)(p + 8);
    return __builtin_shufflevector(lo, hi, 0,1,2,3,4,5,6,7,8,9,10,11,12,13,14,15);
}
__device__ __forceinline__ v16h load8x2(const f16* p0, const f16* p1) {
    v8h lo = *(const v8h*)(p0);
    v8h hi = *(const v8h*)(p1);
    return __builtin_shufflevector(lo, hi, 0,1,2,3,4,5,6,7,8,9,10,11,12,13,14,15);
}
__device__ __forceinline__ v8f wmma_f16(v16h a, v16h b, v8f c) {
    return __builtin_amdgcn_wmma_f32_16x16x32_f16(false, a, false, b, (short)0, c, false, false);
}

// ---------------- prep: f32 -> f16 ----------------
__global__ void cvt_f16_kernel(const float* __restrict__ in, f16* __restrict__ out, int n) {
    int i = blockIdx.x * blockDim.x + threadIdx.x;
    if (i < n) out[i] = (f16)in[i];
}

// ---------------- prep: W[K][N] f32 -> Wt[N][K] f16 ----------------
__global__ void wtrans_kernel(const float* __restrict__ W, f16* __restrict__ Wt, int K, int N) {
    int i = blockIdx.x * blockDim.x + threadIdx.x;
    if (i < K * N) {
        int n = i / K, k = i % K;
        Wt[i] = (f16)W[k * N + n];
    }
}

// ---------------- QKV projection GEMM (WMMA) ----------------
// A:[SQ][DIN] f16, Wt:[AH][DIN] f16, bias:[AH] f32
// transposeOut==0: out[SQ][AH] f16 (pre-scaled)    (Q, K)
// transposeOut==1: out[AH][SQ] f16                 (V transposed)
__global__ __launch_bounds__(256)
void qkv_gemm_kernel(const f16* __restrict__ A, const f16* __restrict__ Wt,
                     const float* __restrict__ bias, f16* __restrict__ out,
                     int transposeOut, float scale) {
    const int lane = threadIdx.x & 31;
    const int wave = threadIdx.x >> 5;
    const int mt   = blockIdx.x;                 // 0..255  (M tiles of 16)
    const int nt   = blockIdx.y * 8 + wave;      // 0..23   (N tiles of 16)
    const int nlo  = lane & 15;
    const int hi8  = (lane < 16) ? 0 : 8;
    const int b0   = (lane < 16) ? 0 : 8;        // A chunk base
    const int c0   = (lane < 16) ? 0 : 16;       // unused; B also uses b0

    const f16* Arow = A  + (size_t)(mt * 16 + nlo) * DIN;
    const f16* Brow = Wt + (size_t)(nt * 16 + nlo) * DIN;

    v8f c = {};
    #pragma unroll 4
    for (int k = 0; k < DIN; k += 32) {
        v16h a = load8x2(Arow + k + b0, Arow + k + b0 + 16);
        v16h b = load8x2(Brow + k + b0, Brow + k + b0 + 16);
        c = wmma_f16(a, b, c);
    }
    const float bv = bias[nt * 16 + nlo];

    if (transposeOut) {
        v8h o;
        #pragma unroll
        for (int r = 0; r < 8; ++r) o[r] = (f16)((c[r] + bv) * scale);
        // row d = nt*16+nlo, cols mt*16 + hi8 + [0..8) are contiguous
        *(v8h*)(out + (size_t)(nt * 16 + nlo) * SQ + mt * 16 + hi8) = o;
    } else {
        #pragma unroll
        for (int r = 0; r < 8; ++r) {
            out[(size_t)(mt * 16 + r + hi8) * AH + nt * 16 + nlo] =
                (f16)((c[r] + bv) * scale);
        }
    }
}

// online-softmax update for one 16-row q tile (two 16x16 score tiles s0|s1),
// writes probabilities (f16) into Prow (row-major 16x32 block, row stride 32)
__device__ __forceinline__ void softmax_update(v8f& s0, v8f& s1, float mk0, float mk1,
                                               float* mr, float* lr, v8f& o0, v8f& o1,
                                               f16* Prow, int nlo, int hi8) {
    #pragma unroll
    for (int r = 0; r < 8; ++r) {
        float a0 = s0[r] + mk0;
        float a1 = s1[r] + mk1;
        float t  = fmaxf(a0, a1);
        t = fmaxf(t, __shfl_xor(t, 1, 32));
        t = fmaxf(t, __shfl_xor(t, 2, 32));
        t = fmaxf(t, __shfl_xor(t, 4, 32));
        t = fmaxf(t, __shfl_xor(t, 8, 32));
        float mn = fmaxf(mr[r], t);
        float al = __expf(mr[r] - mn);
        float p0 = __expf(a0 - mn);
        float p1 = __expf(a1 - mn);
        float rs = p0 + p1;
        rs += __shfl_xor(rs, 1, 32);
        rs += __shfl_xor(rs, 2, 32);
        rs += __shfl_xor(rs, 4, 32);
        rs += __shfl_xor(rs, 8, 32);
        lr[r] = lr[r] * al + rs;
        mr[r] = mn;
        o0[r] *= al;
        o1[r] *= al;
        Prow[(r + hi8) * 32 + nlo]      = (f16)p0;
        Prow[(r + hi8) * 32 + 16 + nlo] = (f16)p1;
    }
}

// ---------------- flash attention (WMMA, one wave per 32 q-rows per head) ----------------
// Qh:[SQ][AH] f16 (pre-scaled by 1/sqrt(HD)), Kh:[SK][AH] f16, Vt:[AH][SK] f16,
// mask:[SK] f32, out:[SQ][AH] f32
__global__ __launch_bounds__(32)
void flash_attn_kernel(const f16* __restrict__ Qh, const f16* __restrict__ Kh,
                       const f16* __restrict__ Vt, const float* __restrict__ mask,
                       float* __restrict__ out) {
    const int lane = threadIdx.x & 31;
    const int qt   = blockIdx.x;      // 0..127 (32 q rows each)
    const int h    = blockIdx.y;      // 0..11
    const int nlo  = lane & 15;
    const int hi8  = (lane < 16) ? 0 : 8;
    const int b0   = (lane < 16) ? 0 : 8;
    const int c0   = (lane < 16) ? 0 : 16;

    __shared__ f16 P[32 * 32];  // two stacked 16x32 probability tiles

    // Preload both Q A-tiles: rows qt*32+[0,16) and qt*32+[16,32)
    const f16* QrowA = Qh + (size_t)(qt * 32 + nlo) * AH + h * HD;
    const f16* QrowB = Qh + (size_t)(qt * 32 + 16 + nlo) * AH + h * HD;
    const v16h aQA = load8x2(QrowA + b0, QrowA + b0 + 16);
    const v16h aQB = load8x2(QrowB + b0, QrowB + b0 + 16);

    v8f oA0 = {}, oA1 = {}, oB0 = {}, oB1 = {};
    float mrA[8], lrA[8], mrB[8], lrB[8];
    #pragma unroll
    for (int r = 0; r < 8; ++r) {
        mrA[r] = -1.0e30f; lrA[r] = 0.0f;
        mrB[r] = -1.0e30f; lrB[r] = 0.0f;
    }

    const f16* Kbase  = Kh + h * HD;
    const f16* Vbase0 = Vt + (size_t)(h * HD + nlo) * SK;       // d in [0,16)
    const f16* Vbase1 = Vt + (size_t)(h * HD + 16 + nlo) * SK;  // d in [16,32)

    for (int kc = 0; kc < SK; kc += 32) {
        // ---- shared B operands for this key chunk
        v16h bK0 = load8x2(Kbase + (size_t)(kc + nlo) * AH + b0, Kbase + (size_t)(kc + nlo) * AH + b0 + 16);
        v16h bK1 = load8x2(Kbase + (size_t)(kc + 16 + nlo) * AH + b0, Kbase + (size_t)(kc + 16 + nlo) * AH + b0 + 16);
        v16h bV0 = load8x2(Vbase0 + kc + b0, Vbase0 + kc + b0 + 16);
        v16h bV1 = load8x2(Vbase1 + kc + b0, Vbase1 + kc + b0 + 16);
        if (kc + 32 < SK) {
            __builtin_prefetch(Kbase + (size_t)(kc + 32 + nlo) * AH + c0, 0, 3);
            __builtin_prefetch(Vbase0 + kc + 32 + c0, 0, 3);
        }

        // ---- S = Q * K^T : four 16x16 tiles (2 q tiles x 2 key halves)
        v8f zero = {};
        v8f sA0 = wmma_f16(aQA, bK0, zero);
        v8f sA1 = wmma_f16(aQA, bK1, zero);
        v8f sB0 = wmma_f16(aQB, bK0, zero);
        v8f sB1 = wmma_f16(aQB, bK1, zero);

        const float mk0 = mask[kc + nlo];
        const float mk1 = mask[kc + 16 + nlo];

        softmax_update(sA0, sA1, mk0, mk1, mrA, lrA, oA0, oA1, P,           nlo, hi8);
        softmax_update(sB0, sB1, mk0, mk1, mrB, lrB, oB0, oB1, P + 16 * 32, nlo, hi8);
        __syncthreads();

        // ---- O += P * V : A tiles reloaded from LDS in A-layout
        v16h aPA = load8x2(&P[nlo * 32 + b0],              &P[nlo * 32 + b0 + 16]);
        v16h aPB = load8x2(&P[(16 + nlo) * 32 + b0],       &P[(16 + nlo) * 32 + b0 + 16]);
        oA0 = wmma_f16(aPA, bV0, oA0);
        oA1 = wmma_f16(aPA, bV1, oA1);
        oB0 = wmma_f16(aPB, bV0, oB0);
        oB1 = wmma_f16(aPB, bV1, oB1);
        __syncthreads();
    }

    // ---- epilogue: O / l, store f32 for both q tiles
    #pragma unroll
    for (int r = 0; r < 8; ++r) {
        const int M = r + hi8;
        float invA = 1.0f / lrA[r];
        float invB = 1.0f / lrB[r];
        float* orowA = out + (size_t)(qt * 32 + M) * AH + h * HD;
        float* orowB = out + (size_t)(qt * 32 + 16 + M) * AH + h * HD;
        orowA[nlo]      = oA0[r] * invA;
        orowA[16 + nlo] = oA1[r] * invA;
        orowB[nlo]      = oB0[r] * invB;
        orowB[16 + nlo] = oB1[r] * invB;
    }
}

// ---------------- host-side launch ----------------
extern "C" void kernel_launch(void* const* d_in, const int* in_sizes, int n_in,
                              void* d_out, int out_size, void* d_ws, size_t ws_size,
                              hipStream_t stream) {
    const float* X    = (const float*)d_in[0];  // hidden_states        [SQ][DIN]
    const float* E    = (const float*)d_in[1];  // encoder_hidden_states[SK][DIN]
    const float* msk  = (const float*)d_in[2];  // mask                 [SK]
    const float* Wq   = (const float*)d_in[3];
    const float* bq   = (const float*)d_in[4];
    const float* Wk   = (const float*)d_in[5];
    const float* bk   = (const float*)d_in[6];
    const float* Wv   = (const float*)d_in[7];
    const float* bv   = (const float*)d_in[8];
    float* out        = (float*)d_out;

    char* ws = (char*)d_ws;
    size_t off = 0;
    auto alloc = [&](size_t bytes) { char* p = ws + off; off = (off + bytes + 255) & ~(size_t)255; return p; };
    f16* Xh  = (f16*)alloc((size_t)SQ * DIN * 2);
    f16* Eh  = (f16*)alloc((size_t)SK * DIN * 2);
    f16* Wtq = (f16*)alloc((size_t)AH * DIN * 2);
    f16* Wtk = (f16*)alloc((size_t)AH * DIN * 2);
    f16* Wtv = (f16*)alloc((size_t)AH * DIN * 2);
    f16* Qh  = (f16*)alloc((size_t)SQ * AH * 2);
    f16* Kh  = (f16*)alloc((size_t)SK * AH * 2);
    f16* Vt  = (f16*)alloc((size_t)AH * SK * 2);

    // 1) convert activations to f16
    {
        int n = SQ * DIN;
        cvt_f16_kernel<<<(n + 255) / 256, 256, 0, stream>>>(X, Xh, n);
        cvt_f16_kernel<<<(n + 255) / 256, 256, 0, stream>>>(E, Eh, n);
    }
    // 2) transpose + convert weights
    {
        int n = DIN * AH;
        wtrans_kernel<<<(n + 255) / 256, 256, 0, stream>>>(Wq, Wtq, DIN, AH);
        wtrans_kernel<<<(n + 255) / 256, 256, 0, stream>>>(Wk, Wtk, DIN, AH);
        wtrans_kernel<<<(n + 255) / 256, 256, 0, stream>>>(Wv, Wtv, DIN, AH);
    }
    // 3) QKV projections (WMMA). Q pre-scaled by 1/sqrt(HEAD_DIM).
    {
        dim3 grid(SQ / 16, (AH / 16) / 8, 1);  // 256 x 3
        dim3 block(256, 1, 1);
        const float qscale = 1.0f / sqrtf((float)HD);
        qkv_gemm_kernel<<<grid, block, 0, stream>>>(Xh, Wtq, bq, Qh, 0, qscale);
        qkv_gemm_kernel<<<grid, block, 0, stream>>>(Eh, Wtk, bk, Kh, 0, 1.0f);
        qkv_gemm_kernel<<<grid, block, 0, stream>>>(Eh, Wtv, bv, Vt, 1, 1.0f);
    }
    // 4) flash attention (WMMA), 32 q rows per wave
    {
        dim3 grid(SQ / 32, NH, 1);  // 128 x 12
        dim3 block(32, 1, 1);
        flash_attn_kernel<<<grid, block, 0, stream>>>(Qh, Kh, Vt, msk, out);
    }
}
